// RoutingNetwork_66967130079473
// MI455X (gfx1250) — compile-verified, matrix operand order checked by reading, not run on hardware
//
#include <hip/hip_runtime.h>
#include <hip/hip_bf16.h>

// MoE router: scores = x @ W^T (M=16384, N=64, K=2048), then top-2 over experts.
// softmax is monotonic -> top-2 of probs == top-2 of raw scores; returned
// weights are the raw scores (stop_gradient, scale 1.0) -> no softmax math.
//
// MI455X roofline: x stream = 134MB / 23.3 TB/s ~= 5.8us floor; reaching it
// needs ~740 TFLOP/s effective fp32 -> must run on the matrix pipe:
// v_wmma_f32_16x16x32_bf16 with fp32 split into bf16 hi/lo
// (hi = bf16(v), lo = bf16(v - hi)); 3 WMMAs/tile gives ~fp32 accuracy at
// bf16-matrix throughput.
//
// Data movement: W (512KB) is pre-packed once per launch into WMMA-fragment-
// ordered bf16 hi/lo records in d_ws, so the hot loop's B path is pure
// global_load_b128 from L2 with zero conversion VALU. Only the HBM-streamed
// A operand (x) is split in-registers (~30 VALU/chunk, v_cvt_pk_bf16_f32 +
// v_pk_add_f32-neg). Loop mix per chunk: 12 wmma, ~20 vmem, ~35 valu.

typedef __attribute__((ext_vector_type(16))) __bf16 v16bf;
typedef __attribute__((ext_vector_type(8)))  float  v8f;

constexpr int TOKENS  = 16384;
constexpr int HIDDEN  = 2048;
constexpr int EXPERTS = 64;
constexpr int CHUNKS  = HIDDEN / 32;                 // 64 K-chunks
constexpr int WAVES_PER_BLOCK  = 4;
constexpr int TOKENS_PER_WAVE  = 16;
constexpr int TOKENS_PER_BLOCK = WAVES_PER_BLOCK * TOKENS_PER_WAVE; // 64

// Packed W: per (chunk, tile, lane) one 64B record = 16 bf16 hi ++ 16 bf16 lo,
// in exact WMMA B-fragment element order. Lane-contiguous -> coalesced.
constexpr size_t PACK_BYTES = (size_t)CHUNKS * 4 * 32 * 64;  // 512 KB in d_ws

// CDNA5 16-bit A-matrix 16x32 layout: lane (m = lane%16, half = lane/16)
// holds, in fragment-element order 0..15, the floats
//   row[kc + 8*half + {0..7, 16..23}]  == four contiguous float4 loads.
// B (32x16 = W^T tile) mirrors this with lane%16 = expert column.
__device__ __forceinline__ void load_chunk16(const float* __restrict__ row,
                                             int kc, int half,
                                             float* __restrict__ f) {
  const float4 f0 = *reinterpret_cast<const float4*>(row + kc + half * 8 + 0);
  const float4 f1 = *reinterpret_cast<const float4*>(row + kc + half * 8 + 4);
  const float4 f2 = *reinterpret_cast<const float4*>(row + kc + half * 8 + 16);
  const float4 f3 = *reinterpret_cast<const float4*>(row + kc + half * 8 + 20);
  f[0]  = f0.x; f[1]  = f0.y; f[2]  = f0.z; f[3]  = f0.w;
  f[4]  = f1.x; f[5]  = f1.y; f[6]  = f1.z; f[7]  = f1.w;
  f[8]  = f2.x; f[9]  = f2.y; f[10] = f2.z; f[11] = f2.w;
  f[12] = f3.x; f[13] = f3.y; f[14] = f3.z; f[15] = f3.w;
}

__device__ __forceinline__ void split_bf16(const float* __restrict__ f,
                                           v16bf& hi, v16bf& lo) {
#pragma unroll
  for (int i = 0; i < 16; ++i) {
    const float v = f[i];
    const __bf16 h = (__bf16)v;
    hi[i] = h;
    lo[i] = (__bf16)(v - (float)h);
  }
}

// ---------------- W pre-pack kernel (once per launch, 512KB out) ------------
__global__ __launch_bounds__(128)
void pack_w_kernel(const float* __restrict__ W, __bf16* __restrict__ pack) {
  const int tid    = blockIdx.x * blockDim.x + threadIdx.x; // 8192 threads
  const int lane   = tid & 31;
  const int t      = (tid >> 5) & 3;
  const int chunk  = tid >> 7;
  const int lane16 = lane & 15;
  const int half   = lane >> 4;

  float f[16];
  load_chunk16(W + (size_t)(t * 16 + lane16) * HIDDEN, chunk * 32, half, f);
  v16bf hi, lo;
  split_bf16(f, hi, lo);

  __bf16* dst = pack + (size_t)tid * 32;            // 64B record per thread
  *reinterpret_cast<v16bf*>(dst + 0)  = hi;
  *reinterpret_cast<v16bf*>(dst + 16) = lo;
}

// ---------------- main GEMM + top-2 kernel ----------------------------------
__global__ __launch_bounds__(WAVES_PER_BLOCK * 32)
void router_topk_kernel(const float* __restrict__ x,
                        const __bf16* __restrict__ pack,
                        float* __restrict__ out) {
  __shared__ float smem[WAVES_PER_BLOCK][TOKENS_PER_WAVE][EXPERTS]; // 16 KB

  const int lane   = threadIdx.x & 31;
  const int wave   = threadIdx.x >> 5;
  const int half   = lane >> 4;
  const int lane16 = lane & 15;

  const int tokBase = blockIdx.x * TOKENS_PER_BLOCK + wave * TOKENS_PER_WAVE;

  const float* __restrict__ xrow = x + (size_t)(tokBase + lane16) * HIDDEN;

  const v8f zero8 = {0.f, 0.f, 0.f, 0.f, 0.f, 0.f, 0.f, 0.f};
  v8f acc[4];
#pragma unroll
  for (int t = 0; t < 4; ++t) acc[t] = zero8;

#pragma unroll 2
  for (int kc = 0; kc < HIDDEN; kc += 32) {
    // Cover HBM latency on the strided x stream (gfx1250 global_prefetch_b8).
    __builtin_prefetch(xrow + kc + 256, 0, 0);

    // ---- A: stream x, split fp32 -> bf16 hi/lo in registers ----
    float fa[16];
    load_chunk16(xrow, kc, half, fa);
    v16bf a_hi, a_lo;
    split_bf16(fa, a_hi, a_lo);

    // ---- B: 4 expert tiles, pre-packed bf16 hi/lo fragments from L2 ----
    v16bf b_hi[4], b_lo[4];
    const __bf16* bp = pack + ((size_t)(kc >> 5) * 4 * 32 + lane) * 32;
#pragma unroll
    for (int t = 0; t < 4; ++t) {
      b_hi[t] = *reinterpret_cast<const v16bf*>(bp + (size_t)t * 32 * 32);
      b_lo[t] = *reinterpret_cast<const v16bf*>(bp + (size_t)t * 32 * 32 + 16);
    }

    // D = A*B + C with fp32-split operands: hi*hi + lo*hi + hi*lo
#pragma unroll
    for (int t = 0; t < 4; ++t) {
      acc[t] = __builtin_amdgcn_wmma_f32_16x16x32_bf16(
          false, a_hi, false, b_hi[t], (short)0, acc[t], false, false);
      acc[t] = __builtin_amdgcn_wmma_f32_16x16x32_bf16(
          false, a_lo, false, b_hi[t], (short)0, acc[t], false, false);
      acc[t] = __builtin_amdgcn_wmma_f32_16x16x32_bf16(
          false, a_hi, false, b_lo[t], (short)0, acc[t], false, false);
    }
  }

  // C/D layout: VGPR j, lane l -> row (j + 8*(l/16)), col l%16.
#pragma unroll
  for (int t = 0; t < 4; ++t)
#pragma unroll
    for (int j = 0; j < 8; ++j)
      smem[wave][8 * half + j][t * 16 + lane16] = acc[t][j];

  __syncthreads();

  // Top-2 scan: 16 lanes/wave, one token each, 64 contiguous LDS floats.
  // Strict '>' gives lower-index-wins tie-breaking (matches lax.top_k).
  if (lane < TOKENS_PER_WAVE) {
    const float* __restrict__ s = smem[wave][lane];
    float best = -3.402823466e38f, second = -3.402823466e38f;
    int bi = 0, si = 0;
#pragma unroll 8
    for (int e = 0; e < EXPERTS; ++e) {
      const float v = s[e];
      if (v > best) {
        second = best; si = bi;
        best   = v;    bi = e;
      } else if (v > second) {
        second = v; si = e;
      }
    }
    const int tok = tokBase + lane;
    out[(size_t)tok * 2 + 0] = best;    // routing weight = raw score
    out[(size_t)tok * 2 + 1] = second;
    out[(size_t)TOKENS * 2 + (size_t)tok * 2 + 0] = (float)bi;
    out[(size_t)TOKENS * 2 + (size_t)tok * 2 + 1] = (float)si;
  }
}

extern "C" void kernel_launch(void* const* d_in, const int* in_sizes, int n_in,
                              void* d_out, int out_size, void* d_ws, size_t ws_size,
                              hipStream_t stream) {
  const float* x = (const float*)d_in[0];   // [16384, 2048] fp32
  const float* W = (const float*)d_in[1];   // [64, 2048] fp32
  float* out = (float*)d_out;               // 32768 weights ++ 32768 indices
  __bf16* pack = reinterpret_cast<__bf16*>(d_ws);  // needs 512 KB (ws_size)

  (void)in_sizes; (void)n_in; (void)out_size; (void)ws_size;

  pack_w_kernel<<<dim3(CHUNKS), dim3(128), 0, stream>>>(W, pack);

  dim3 grid(TOKENS / TOKENS_PER_BLOCK);     // 256 workgroups
  dim3 block(WAVES_PER_BLOCK * 32);         // 128 threads = 4 waves (wave32)
  router_topk_kernel<<<grid, block, 0, stream>>>(x, pack, out);
}
